// PreBlock_31078383353957
// MI455X (gfx1250) — matrix operand order repaired and checked by reading, not compiled, not run on hardware
//
#include <hip/hip_runtime.h>

typedef __attribute__((ext_vector_type(16))) _Float16 v16h;
typedef __attribute__((ext_vector_type(8)))  float    v8f;
typedef __attribute__((ext_vector_type(4)))  int      v4i;

namespace {
constexpr int SZ  = 128;           // spatial size
constexpr int NB  = 16;            // batch
constexpr int GCh = 32;            // growth / output channels per conv
constexpr int FC  = 160;           // fea channels (32 + 4*32)
constexpr int SS  = SZ * SZ;
}

// CDNA5 async global->LDS copy path (probe via __has_builtin; safe fallback).
#if __has_builtin(__builtin_amdgcn_global_load_async_to_lds_b128) && \
    __has_builtin(__builtin_amdgcn_s_wait_asynccnt)
#define USE_ASYNC_COPY 1
typedef __attribute__((address_space(1))) v4i* async_g128_t;   // global src
typedef __attribute__((address_space(3))) v4i* async_l128_t;   // LDS dst
#else
#define USE_ASYNC_COPY 0
#endif

// ---------------------------------------------------------------------------
// Pack conv weights [32, IC, 3, 3] (f32) into WMMA 16x32 f16 A-fragments.
// K-order: k = r*IC + ic  (tap r = kh*3+kw outer, channel inner).  Because
// IC % 32 == 0, every 32-wide k-chunk has a single (kh,kw) and contiguous ic.
// A-layout (16-bit A 16x32, wave32): lane = hi*16+lh holds row M=lh,
// element i -> K = (i&7) + 8*hi + 16*(i>>3).
// ---------------------------------------------------------------------------
__global__ void pack_conv_a(const float* __restrict__ w,
                            _Float16* __restrict__ ap, int IC, int KC) {
    int idx = blockIdx.x * blockDim.x + threadIdx.x;
    int total = 2 * KC * 512;                 // 2 mtiles * KC * 32 lanes * 16
    if (idx >= total) return;
    int i    = idx & 15;
    int lane = (idx >> 4) & 31;
    int kc   = (idx >> 9) % KC;
    int mt   = idx / (KC * 512);
    int hi = lane >> 4, lh = lane & 15;
    int k  = kc * 32 + (i & 7) + 8 * hi + 16 * (i >> 3);
    int oc = mt * 16 + lh;
    float v = 0.0f;
    if (k < IC * 9) {
        int r  = k / IC;                      // tap index 0..8
        int ic = k - r * IC;                  // channel
        v = w[(oc * IC + ic) * 9 + r];        // [OC,IC,3,3] flat: tap is inner
    }
    ap[idx] = (_Float16)v;
}

// ---------------------------------------------------------------------------
// Pack H (128x128 f32) into:
//   ha: A-fragments of H   (for T = H * X)
//   hb: B-fragments of H^T (for Y = T * H^T): b[i] = H[n][k], k = i + 16*hi
// ---------------------------------------------------------------------------
__global__ void pack_h(const float* __restrict__ H,
                       _Float16* __restrict__ ha, _Float16* __restrict__ hb) {
    int idx = blockIdx.x * blockDim.x + threadIdx.x;
    if (idx >= 8 * 4 * 32 * 16) return;
    int i    = idx & 15;
    int lane = (idx >> 4) & 31;
    int kc   = (idx >> 9) & 3;
    int t    = idx >> 11;                     // tile (mt for ha, nt for hb)
    int hi = lane >> 4, lh = lane & 15;
    int kA = kc * 32 + (i & 7) + 8 * hi + 16 * (i >> 3);
    int kB = kc * 32 + i + 16 * hi;
    ha[idx] = (_Float16)H[(t * 16 + lh) * SZ + kA];
    hb[idx] = (_Float16)H[(t * 16 + lh) * SZ + kB];
}

// ---------------------------------------------------------------------------
// Copy x [B,32,S,S] into channels 0..31 of fea [B,160,S,S] (no ReLU: convs
// 2..4 consume the original x; ReLU happens inside the DCT kernel).
// ---------------------------------------------------------------------------
__global__ void copy_x_to_fea(const float* __restrict__ x,
                              float* __restrict__ fea) {
    int idx = blockIdx.x * blockDim.x + threadIdx.x;
    if (idx >= NB * GCh * SS) return;
    int sp = idx % SS;
    int c  = (idx / SS) % GCh;
    int b  = idx / (GCh * SS);
    fea[((size_t)b * FC + c) * SS + sp] = x[idx];
}

// ---------------------------------------------------------------------------
// Implicit-GEMM dilated 3x3 conv via WMMA, pad == dil.
// K-order r*IC+ic: each 32-wide chunk has constant (kh,kw) -> the 16
// B-fragment loads per chunk are one base address + ((i&7)+16*(i>>3))*SS immediate offsets
// (unconditional, clamped, clause-friendly; OOB handled by one select mask).
// Block: 128 threads = 4 waves = (2 oc-tiles) x (2 w-tiles of 16).
// Grid: (SZ/32, SZ, NB).
// ---------------------------------------------------------------------------
__global__ __launch_bounds__(128) void conv3x3_wmma(
    const float* __restrict__ in, const _Float16* __restrict__ ap,
    const float* __restrict__ bias, const float* __restrict__ res,
    float* __restrict__ out, int IC, int KC, int dil, int outCh, int doRelu) {
    const int lane = threadIdx.x & 31;
    const int wv   = threadIdx.x >> 5;
    const int mt   = wv >> 1, nt = wv & 1;
    const int lh   = lane & 15, hi = lane >> 4;
    const int h    = blockIdx.y;
    const int b    = blockIdx.z;
    const int w0   = blockIdx.x * 32 + nt * 16 + lh;  // this lane's column

    const float* inB = in + (size_t)b * FC * SS;

    v8f acc = {0.f, 0.f, 0.f, 0.f, 0.f, 0.f, 0.f, 0.f};
    int r = 0, icBase = 0;                    // chunk-uniform K bookkeeping
    for (int kc = 0; kc < KC; ++kc) {
        v16h a = *(const v16h*)(ap + ((size_t)(mt * KC + kc) * 32 + lane) * 16);
        if (kc + 1 < KC)
            __builtin_prefetch(ap + ((size_t)(mt * KC + kc + 1) * 32 + lane) * 16, 0, 3);

        int kh = r / 3;                       // constant-divisor, chunk-uniform
        int kw = r - kh * 3;
        int hh = h + dil * (kh - 1);          // scalar per chunk
        int ww = w0 + dil * (kw - 1);         // per lane
        bool valid = (hh >= 0) & (hh < SZ) & (ww >= 0) & (ww < SZ);
        int hc = min(max(hh, 0), SZ - 1);
        int wc = min(max(ww, 0), SZ - 1);
        const float* p = inB + ((size_t)(icBase + 8 * hi) * SZ + hc) * SZ + wc;

        v16h bf;
#pragma unroll
        for (int i = 0; i < 16; ++i) {        // 16 loads, immediate offsets
            float v = p[(size_t)((i & 7) + 16 * (i >> 3)) * SS];
            bf[i] = valid ? (_Float16)v : (_Float16)0.f;
        }
        acc = __builtin_amdgcn_wmma_f32_16x16x32_f16(false, a, false, bf,
                                                     (short)0, acc, false, false);
        icBase += 32;
        if (icBase == IC) { icBase = 0; ++r; }
    }
#pragma unroll
    for (int j = 0; j < 8; ++j) {
        int oc   = mt * 16 + j + 8 * hi;      // D-layout: M = j + 8*hi
        float v  = acc[j] + bias[oc];
        if (doRelu) v = fmaxf(v, 0.f);
        size_t oidx = ((size_t)b * outCh + oc) * SS + (size_t)h * SZ + w0;
        if (res) v += res[oidx];
        out[oidx] = v;
    }
}

// ---------------------------------------------------------------------------
// Fused two-sided DCT per (b,c) image: Y = H * relu(X) * H^T, in place.
// One block (8 waves) per image.  X staged raw f32 in 64KB LDS (async
// global->LDS copy on CDNA5 when available); ReLU + f16 convert at fragment
// build.  After MM1 the first 32KB is reused (XOR bank swizzle) to stage T
// rows as f16; wave m reads back only rows it wrote -> one barrier total.
// ---------------------------------------------------------------------------
__global__ __launch_bounds__(256) void dct_wmma(
    float* __restrict__ fea, const _Float16* __restrict__ ha,
    const _Float16* __restrict__ hb) {
    __shared__ float smemX[SS];               // 64KB: X (f32), then T (f16)
    _Float16* smemT = (_Float16*)smemX;
    const int bc   = blockIdx.x;
    float* X       = fea + (size_t)bc * SS;
    const int t    = threadIdx.x;
    const int lane = t & 31;
    const int mt   = t >> 5;                  // wave id = M-tile 0..7
    const int lh   = lane & 15, hi = lane >> 4;

#if USE_ASYNC_COPY
    for (int e = t * 4; e < SS; e += 256 * 4)
        __builtin_amdgcn_global_load_async_to_lds_b128(
            (async_g128_t)(X + e), (async_l128_t)(smemX + e), 0, 0);
    __builtin_amdgcn_s_wait_asynccnt(0);
#else
    for (int e = t; e < SS; e += 256)
        smemX[e] = X[e];
#endif
    __syncthreads();

    // ---- MM1: T[16mt..16mt+15][:] = H_rows * relu(X) ----
    v8f acc[8];
#pragma unroll
    for (int q = 0; q < 8; ++q) acc[q] = (v8f){0.f,0.f,0.f,0.f,0.f,0.f,0.f,0.f};
    for (int kc = 0; kc < 4; ++kc) {
        v16h a = *(const v16h*)(ha + ((size_t)(mt * 4 + kc) * 32 + lane) * 16);
#pragma unroll
        for (int nt = 0; nt < 8; ++nt) {
            v16h bf;
#pragma unroll
            for (int i = 0; i < 16; ++i) {
                int k = kc * 32 + (i & 7) + 8 * hi + 16 * (i >> 3); // K row of X
                bf[i] = (_Float16)fmaxf(smemX[k * SZ + nt * 16 + lh], 0.f);
            }
            acc[nt] = __builtin_amdgcn_wmma_f32_16x16x32_f16(
                false, a, false, bf, (short)0, acc[nt], false, false);
        }
    }
    __syncthreads();                          // everyone done reading X

    // ---- stage T rows as f16, XOR swizzle: (row,k) -> row*SZ + (k ^ 2*(row&15))
#pragma unroll
    for (int nt = 0; nt < 8; ++nt)
#pragma unroll
        for (int j = 0; j < 8; ++j) {
            int row = mt * 16 + j + 8 * hi;
            int col = nt * 16 + lh;
            smemT[row * SZ + (col ^ ((row & 15) << 1))] = (_Float16)acc[nt][j];
        }

    // ---- MM2: Y[16mt..16mt+15][:] = T_rows * H^T ----
    v8f acc2[8];
#pragma unroll
    for (int q = 0; q < 8; ++q) acc2[q] = (v8f){0.f,0.f,0.f,0.f,0.f,0.f,0.f,0.f};
    for (int kc = 0; kc < 4; ++kc) {
        v16h a;
        int row = mt * 16 + lh;               // A rows: M = lh for both halves
#pragma unroll
        for (int i = 0; i < 16; ++i) {
            int k = kc * 32 + i + 16 * hi;
            a[i] = smemT[row * SZ + (k ^ ((row & 15) << 1))];
        }
#pragma unroll
        for (int nt = 0; nt < 8; ++nt) {
            v16h bf = *(const v16h*)(hb + ((size_t)(nt * 4 + kc) * 32 + lane) * 16);
            acc2[nt] = __builtin_amdgcn_wmma_f32_16x16x32_f16(
                false, a, false, bf, (short)0, acc2[nt], false, false);
        }
    }
#pragma unroll
    for (int nt = 0; nt < 8; ++nt)
#pragma unroll
        for (int j = 0; j < 8; ++j) {
            int row = mt * 16 + j + 8 * hi;
            int col = nt * 16 + lh;
            X[row * SZ + col] = acc2[nt][j];  // in-place Y
        }
}

// ---------------------------------------------------------------------------
extern "C" void kernel_launch(void* const* d_in, const int* in_sizes, int n_in,
                              void* d_out, int out_size, void* d_ws, size_t ws_size,
                              hipStream_t stream) {
    (void)in_sizes; (void)n_in; (void)out_size; (void)ws_size;
    const float* x  = (const float*)d_in[0];
    const float* w1 = (const float*)d_in[1];
    const float* b1 = (const float*)d_in[2];
    const float* w2 = (const float*)d_in[3];
    const float* b2 = (const float*)d_in[4];
    const float* w3 = (const float*)d_in[5];
    const float* b3 = (const float*)d_in[6];
    const float* w4 = (const float*)d_in[7];
    const float* b4 = (const float*)d_in[8];
    // d_in[9]=w5, d_in[10]=b5: unused by the reference forward
    const float* w6 = (const float*)d_in[11];
    const float* b6 = (const float*)d_in[12];
    const float* Hm = (const float*)d_in[13];
    float* out = (float*)d_out;

    char* ws = (char*)d_ws;
    const size_t feaBytes = (size_t)NB * FC * SS * sizeof(float);  // 167.8 MB
    float*    fea = (float*)ws;
    _Float16* A1  = (_Float16*)(ws + feaBytes);
    _Float16* A2  = A1 + (size_t)2 * 9  * 512;
    _Float16* A3  = A2 + (size_t)2 * 18 * 512;
    _Float16* A4  = A3 + (size_t)2 * 27 * 512;
    _Float16* A6  = A4 + (size_t)2 * 36 * 512;
    _Float16* HA  = A6 + (size_t)2 * 45 * 512;
    _Float16* HB  = HA + 16384;

    // --- pack fragments (cheap, deterministic, every call) ---
    auto packConv = [&](const float* w, _Float16* ap, int IC) {
        int KC = IC * 9 / 32;
        int total = 2 * KC * 512;
        pack_conv_a<<<(total + 255) / 256, 256, 0, stream>>>(w, ap, IC, KC);
    };
    packConv(w1, A1, 32);
    packConv(w2, A2, 64);
    packConv(w3, A3, 96);
    packConv(w4, A4, 128);
    packConv(w6, A6, 160);
    pack_h<<<16384 / 256, 256, 0, stream>>>(Hm, HA, HB);

    // --- fea[:,0:32] = x ---
    {
        int total = NB * GCh * SS;
        copy_x_to_fea<<<(total + 255) / 256, 256, 0, stream>>>(x, fea);
    }

    // --- dense block: 4 dilated convs, outputs appended as channel slices ---
    dim3 cgrid(SZ / 32, SZ, NB);
    conv3x3_wmma<<<cgrid, 128, 0, stream>>>(fea, A1, b1, nullptr,
                                            fea + (size_t)32 * SS, 32, 9, 2, FC, 1);
    conv3x3_wmma<<<cgrid, 128, 0, stream>>>(fea, A2, b2, nullptr,
                                            fea + (size_t)64 * SS, 64, 18, 2, FC, 1);
    conv3x3_wmma<<<cgrid, 128, 0, stream>>>(fea, A3, b3, nullptr,
                                            fea + (size_t)96 * SS, 96, 27, 2, FC, 1);
    conv3x3_wmma<<<cgrid, 128, 0, stream>>>(fea, A4, b4, nullptr,
                                            fea + (size_t)128 * SS, 128, 36, 2, FC, 1);

    // --- fused per-image DCT (applies ReLU internally), in place ---
    dct_wmma<<<NB * FC, 256, 0, stream>>>(fea, HA, HB);

    // --- final conv (pad=1, dil=1) + bias + residual x -> d_out ---
    conv3x3_wmma<<<cgrid, 128, 0, stream>>>(fea, A6, b6, x,
                                            out, 160, 45, 1, GCh, 0);
}
